// Inter_Classifier_12601434047059
// MI455X (gfx1250) — matrix operand order repaired and checked by reading, not compiled, not run on hardware
//
#include <hip/hip_runtime.h>
#include <hip/hip_bf16.h>
#include <math.h>

// ---------------------------------------------------------------------------
// Types for CDNA5 WMMA (gfx1250, wave32)
// ---------------------------------------------------------------------------
typedef __bf16 v16bf __attribute__((ext_vector_type(16)));
typedef __bf16 v8bf  __attribute__((ext_vector_type(8)));
typedef float  v8f   __attribute__((ext_vector_type(8)));

#define BATCH 16384
#define EPS 1e-8f

// round-to-nearest-even f32 -> bf16 (as raw u16)
__device__ __forceinline__ unsigned short f2bf(float f) {
    unsigned int u = __builtin_bit_cast(unsigned int, f);
    u += 0x7FFFu + ((u >> 16) & 1u);
    return (unsigned short)(u >> 16);
}

// A-fragment (16x32 bf16, row-major LDS tile): per 16-bit-A layout,
// lane half h: K(j) = (j<8 ? j : j+8) + 8*h  -> two contiguous 16B LDS loads.
__device__ __forceinline__ v16bf load_a_lds(const unsigned short* rowp, int k0, int h) {
    const v8bf lo = *(const v8bf*)(rowp + k0 + 8 * h);
    const v8bf hi = *(const v8bf*)(rowp + k0 + 16 + 8 * h);
    return __builtin_shufflevector(lo, hi, 0,1,2,3,4,5,6,7,8,9,10,11,12,13,14,15);
}

// load 4 B fragments (one per N tile) at K offset k
__device__ __forceinline__ void load_b4(v16bf (&b)[4],
                                        const unsigned short* const (&wp)[4], int k) {
#pragma unroll
    for (int t = 0; t < 4; ++t) {
        const v8bf lo = *(const v8bf*)(wp[t] + k);
        const v8bf hi = *(const v8bf*)(wp[t] + k + 16);
        b[t] = __builtin_shufflevector(lo, hi, 0,1,2,3,4,5,6,7,8,9,10,11,12,13,14,15);
    }
}

// 4 WMMAs sharing one A fragment
__device__ __forceinline__ void wmma4(v8f (&acc)[4], v16bf a, const v16bf (&b)[4]) {
#pragma unroll
    for (int t = 0; t < 4; ++t)
        acc[t] = __builtin_amdgcn_wmma_f32_16x16x32_bf16(
            false, a, false, b[t], (short)0, acc[t], false, false);
}

// ---------------------------------------------------------------------------
// K0: convert fp32 weight [K][N] -> bf16 transposed [N][K]
// ---------------------------------------------------------------------------
__global__ void convert_transpose_kernel(const float* __restrict__ W,
                                         unsigned short* __restrict__ WT,
                                         int K, int N) {
    int idx = blockIdx.x * blockDim.x + threadIdx.x;
    if (idx < K * N) {
        int k = idx / N, n = idx % N;
        WT[(size_t)n * K + k] = f2bf(W[idx]);
    }
}

// ---------------------------------------------------------------------------
// K1: fused 2-layer MLP projection.  Z = relu(X@W1+b1)@W2+b2
//   X: [rows, IN] fp32,  W1T: [HID][IN] bf16,  W2T: [256][HID] bf16
//   block = 128 threads (4 waves), one 16-row M tile per block.
//   Ping-pong B prefetch + FULL K unroll: unique registers per load (no
//   buffer-rotation copies) with loads issued one K-step ahead of use.
// ---------------------------------------------------------------------------
template<int IN, int HID>
__global__ __launch_bounds__(128)
void proj_kernel(const float* __restrict__ X,
                 const unsigned short* __restrict__ W1T,
                 const float* __restrict__ b1,
                 const unsigned short* __restrict__ W2T,
                 const float* __restrict__ b2,
                 float* __restrict__ Z) {
    constexpr int LDX = IN + 8;    // bf16 elems; row stride 16B-aligned
    constexpr int LDH = HID + 8;
    __shared__ __attribute__((aligned(32))) unsigned short Xt[16 * LDX];
    __shared__ __attribute__((aligned(32))) unsigned short Ht[16 * LDH];

    const int tid  = threadIdx.x;
    const int wave = tid >> 5;
    const int lane = tid & 31;
    const int m    = lane & 15;    // row (A) / col-in-tile (B, C/D)
    const int h    = lane >> 4;    // lane half
    const int row0 = blockIdx.x * 16;

    // ---- Phase 1: stage X tile (fp32 -> bf16) into LDS --------------------
    constexpr int VEC = IN / 4;
    for (int idx = tid; idx < 16 * VEC; idx += 128) {
        int r = idx / VEC, c = (idx % VEC) * 4;
        const float4 v = *(const float4*)(X + (size_t)(row0 + r) * IN + c);
        unsigned short* d = Xt + r * LDX + c;
        d[0] = f2bf(v.x); d[1] = f2bf(v.y); d[2] = f2bf(v.z); d[3] = f2bf(v.w);
    }
    __syncthreads();

    // ---- Phase 2: H = relu(X @ W1 + b1), bf16 into LDS --------------------
    constexpr int NT1     = HID / 16;   // hidden N tiles
    constexpr int TPW     = NT1 / 4;    // tiles per wave
    constexpr int KSTEPS1 = IN / 32;    // even (24 or 8)
    for (int g = 0; g < TPW; g += 4) {
        v8f acc[4] = {};
        const int nbase = (wave * TPW + g) * 16;
        const unsigned short* wp[4];
#pragma unroll
        for (int t = 0; t < 4; ++t)
            wp[t] = W1T + (size_t)(nbase + t * 16 + m) * IN + 8 * h;

        v16bf b0[4], b1v[4];
        load_b4(b0, wp, 0);                               // prime pipeline
#pragma unroll
        for (int ks = 0; ks + 2 < KSTEPS1; ks += 2) {
            load_b4(b1v, wp, (ks + 1) * 32);
            wmma4(acc, load_a_lds(Xt + m * LDX, ks * 32, h), b0);
            load_b4(b0, wp, (ks + 2) * 32);
            wmma4(acc, load_a_lds(Xt + m * LDX, (ks + 1) * 32, h), b1v);
        }
        // tail: ks == KSTEPS1 - 2
        load_b4(b1v, wp, (KSTEPS1 - 1) * 32);
        wmma4(acc, load_a_lds(Xt + m * LDX, (KSTEPS1 - 2) * 32, h), b0);
        wmma4(acc, load_a_lds(Xt + m * LDX, (KSTEPS1 - 1) * 32, h), b1v);

#pragma unroll
        for (int t = 0; t < 4; ++t) {
            const int n0 = nbase + t * 16;
            const float bias = b1[n0 + m];
#pragma unroll
            for (int r = 0; r < 8; ++r) {          // C/D layout: row = r + 8h
                float v = acc[t][r] + bias;
                v = v > 0.f ? v : 0.f;
                Ht[(r + 8 * h) * LDH + n0 + m] = f2bf(v);
            }
        }
    }

    // ---- Phase 3: Z = H @ W2 + b2  (N = 256) ------------------------------
    // First B fragments issued BEFORE the barrier (global-only -> overlaps
    // the phase-2 tail of other waves).
    constexpr int KSTEPS2 = HID / 32;              // even (24 or 8)
    const int nbase2 = wave * 64;                  // 4 tiles of 16 per wave
    const unsigned short* wp2[4];
#pragma unroll
    for (int t = 0; t < 4; ++t)
        wp2[t] = W2T + (size_t)(nbase2 + t * 16 + m) * HID + 8 * h;
    v16bf c0[4], c1[4];
    load_b4(c0, wp2, 0);

    __syncthreads();

    {
        v8f acc[4] = {};
#pragma unroll
        for (int ks = 0; ks + 2 < KSTEPS2; ks += 2) {
            load_b4(c1, wp2, (ks + 1) * 32);
            wmma4(acc, load_a_lds(Ht + m * LDH, ks * 32, h), c0);
            load_b4(c0, wp2, (ks + 2) * 32);
            wmma4(acc, load_a_lds(Ht + m * LDH, (ks + 1) * 32, h), c1);
        }
        load_b4(c1, wp2, (KSTEPS2 - 1) * 32);
        wmma4(acc, load_a_lds(Ht + m * LDH, (KSTEPS2 - 2) * 32, h), c0);
        wmma4(acc, load_a_lds(Ht + m * LDH, (KSTEPS2 - 1) * 32, h), c1);

#pragma unroll
        for (int t = 0; t < 4; ++t) {
            const int n0 = nbase2 + t * 16;
            const float bias = b2[n0 + m];
#pragma unroll
            for (int r = 0; r < 8; ++r)
                Z[(size_t)(row0 + r + 8 * h) * 256 + n0 + m] = acc[t][r] + bias;
        }
    }
}

// ---------------------------------------------------------------------------
// K2: branch — 9 cosine pairs per (example, branch); one wave32 each.
// Positive rows cached in registers; only permuted slots load from global
// under wave-uniform branches.
// ---------------------------------------------------------------------------
__global__ __launch_bounds__(256)
void branch_kernel(const float* __restrict__ ht, const float* __restrict__ tt,
                   const float* __restrict__ hs, const float* __restrict__ ts,
                   const int* __restrict__ perms,      // [2][8][BATCH]
                   float* __restrict__ psum, float* __restrict__ nsum) {
    const int gw     = (blockIdx.x * blockDim.x + threadIdx.x) >> 5;
    const int lane   = threadIdx.x & 31;
    const int b      = gw >> 1;
    const int branch = gw & 1;
    if (b >= BATCH) return;

    const float* t2 = (branch == 0) ? ht : tt;
    const float* s2 = (branch == 0) ? ts : hs;
    const int*   p  = perms + branch * 8 * BATCH;

    int prow[8];
#pragma unroll
    for (int i = 0; i < 8; ++i) prow[i] = p[i * BATCH + b];

    // cache the positive rows: element i -> flat index lane + 32*i (slot = i/8)
    float tpv[24], spv[24];
#pragma unroll
    for (int i = 0; i < 24; ++i) {
        tpv[i] = t2[(size_t)b * 768 + lane + 32 * i];
        spv[i] = s2[(size_t)b * 768 + lane + 32 * i];
    }

    float pos = 0.f, neg = 0.f;
#pragma unroll 1
    for (int pair = 0; pair < 9; ++pair) {
        int tr[3] = {b, b, b}, sr[3] = {b, b, b};
        switch (pair) {
            case 1: tr[2] = sr[2] = prow[0]; break;                    // swap tail slot
            case 2: tr[0] = sr[0] = prow[1]; break;                    // swap head slot
            case 3: tr[2] = sr[2] = prow[2]; break;
            case 4: tr[0] = sr[0] = prow[3]; break;
            case 5: tr[0] = tr[1] = tr[2] = prow[4]; break;            // shuffled text
            case 6: sr[0] = sr[1] = sr[2] = prow[5]; break;            // shuffled struc
            case 7: tr[0] = tr[1] = tr[2] = prow[6]; break;
            case 8: sr[0] = sr[1] = sr[2] = prow[7]; break;
            default: break;                                            // pair 0 = positive
        }
        float dot = 0.f, na = 0.f, nb = 0.f;
#pragma unroll
        for (int s = 0; s < 3; ++s) {
            float av[8], cv[8];
            if (tr[s] == b) {
#pragma unroll
                for (int j = 0; j < 8; ++j) av[j] = tpv[8 * s + j];
            } else {
                const float* tp = t2 + (size_t)tr[s] * 768 + s * 256 + lane;
#pragma unroll
                for (int j = 0; j < 8; ++j) av[j] = tp[32 * j];
            }
            if (sr[s] == b) {
#pragma unroll
                for (int j = 0; j < 8; ++j) cv[j] = spv[8 * s + j];
            } else {
                const float* sp = s2 + (size_t)sr[s] * 768 + s * 256 + lane;
#pragma unroll
                for (int j = 0; j < 8; ++j) cv[j] = sp[32 * j];
            }
#pragma unroll
            for (int j = 0; j < 8; ++j) {
                dot += av[j] * cv[j];
                na  += av[j] * av[j];
                nb  += cv[j] * cv[j];
            }
        }
#pragma unroll
        for (int off = 16; off > 0; off >>= 1) {
            dot += __shfl_xor(dot, off, 32);
            na  += __shfl_xor(na,  off, 32);
            nb  += __shfl_xor(nb,  off, 32);
        }
        const float cs = dot / (fmaxf(sqrtf(na), EPS) * fmaxf(sqrtf(nb), EPS));
        const float ex = expf(cs);
        if (pair == 0) pos = ex; else neg += ex;
    }
    if (lane == 0) {
        psum[branch * BATCH + b] = pos;
        nsum[branch * BATCH + b] = neg;
    }
}

// ---------------------------------------------------------------------------
// K3: per-example score + mean loss (atomicAdd into d_out[0])
// ---------------------------------------------------------------------------
__global__ void init_loss_kernel(float* loss) { if (threadIdx.x == 0) *loss = 0.f; }

__global__ __launch_bounds__(256)
void finalize_kernel(const float* __restrict__ psum, const float* __restrict__ nsum,
                     const float* __restrict__ conf,
                     float* __restrict__ score, float* __restrict__ loss) {
    __shared__ float red[256];
    const int idx = blockIdx.x * blockDim.x + threadIdx.x;
    float part = 0.f;
    if (idx < BATCH) {
        const float p0 = psum[idx],         n0 = nsum[idx];
        const float p1 = psum[BATCH + idx], n1 = nsum[BATCH + idx];
        score[idx] = -(p0 + p1);
        part = -conf[idx] * (logf(p0 / (p0 + n0)) + logf(p1 / (p1 + n1)));
    }
    red[threadIdx.x] = part;
    __syncthreads();
    for (int s = 128; s > 0; s >>= 1) {
        if (threadIdx.x < s) red[threadIdx.x] += red[threadIdx.x + s];
        __syncthreads();
    }
    if (threadIdx.x == 0) atomicAdd(loss, red[0] / (float)BATCH);
}

// ---------------------------------------------------------------------------
// K4: logits in-place add (lt holds ht, ls holds hs)
// ---------------------------------------------------------------------------
__global__ void logits_kernel(const float* __restrict__ tt, const float* __restrict__ ts,
                              float* __restrict__ lt, float* __restrict__ ls) {
    const size_t total = (size_t)BATCH * 768;
    for (size_t i = (size_t)blockIdx.x * blockDim.x + threadIdx.x; i < total;
         i += (size_t)gridDim.x * blockDim.x) {
        lt[i] += tt[i];
        ls[i] += ts[i];
    }
}

// ---------------------------------------------------------------------------
// Host launch
// ---------------------------------------------------------------------------
extern "C" void kernel_launch(void* const* d_in, const int* in_sizes, int n_in,
                              void* d_out, int out_size, void* d_ws, size_t ws_size,
                              hipStream_t stream) {
    const float* head_text  = (const float*)d_in[0];
    const float* tail_text  = (const float*)d_in[1];
    const float* head_struc = (const float*)d_in[2];
    const float* tail_struc = (const float*)d_in[3];
    const float* confidence = (const float*)d_in[4];
    const float* Wt1 = (const float*)d_in[5];
    const float* bt1 = (const float*)d_in[6];
    const float* Wt2 = (const float*)d_in[7];
    const float* bt2 = (const float*)d_in[8];
    const float* Ws1 = (const float*)d_in[9];
    const float* bs1 = (const float*)d_in[10];
    const float* Ws2 = (const float*)d_in[11];
    const float* bs2 = (const float*)d_in[12];
    const int*   perms = (const int*)d_in[13];

    // output layout: [loss(1), score(B), logits_text(B*768), logits_struc(B*768)]
    float* out   = (float*)d_out;
    float* loss  = out;
    float* score = out + 1;
    float* lt    = out + 1 + BATCH;                // also serves as ht storage
    float* ls    = lt + (size_t)BATCH * 768;       // also serves as hs storage

    // workspace carve-out (256B aligned chunks)
    size_t off = 0;
    auto carve = [&](size_t bytes) {
        size_t cur = off;
        off += (bytes + 255) & ~(size_t)255;
        return (char*)d_ws + cur;
    };
    unsigned short* wt1T = (unsigned short*)carve((size_t)768 * 768 * 2);
    unsigned short* wt2T = (unsigned short*)carve((size_t)256 * 768 * 2);
    unsigned short* ws1T = (unsigned short*)carve((size_t)256 * 256 * 2);
    unsigned short* ws2T = (unsigned short*)carve((size_t)256 * 256 * 2);
    float* tt   = (float*)carve((size_t)BATCH * 768 * 4);
    float* ts   = (float*)carve((size_t)BATCH * 768 * 4);
    float* psum = (float*)carve((size_t)2 * BATCH * 4);
    float* nsum = (float*)carve((size_t)2 * BATCH * 4);

    init_loss_kernel<<<1, 32, 0, stream>>>(loss);

    convert_transpose_kernel<<<(768 * 768 + 255) / 256, 256, 0, stream>>>(Wt1, wt1T, 768, 768);
    convert_transpose_kernel<<<(768 * 256 + 255) / 256, 256, 0, stream>>>(Wt2, wt2T, 768, 256);
    convert_transpose_kernel<<<(256 * 256 + 255) / 256, 256, 0, stream>>>(Ws1, ws1T, 256, 256);
    convert_transpose_kernel<<<(256 * 256 + 255) / 256, 256, 0, stream>>>(Ws2, ws2T, 256, 256);

    const int rows = BATCH * 3;                    // 49152 -> 3072 blocks of 16 rows
    proj_kernel<768, 768><<<rows / 16, 128, 0, stream>>>(head_text,  wt1T, bt1, wt2T, bt2, lt);
    proj_kernel<768, 768><<<rows / 16, 128, 0, stream>>>(tail_text,  wt1T, bt1, wt2T, bt2, tt);
    proj_kernel<256, 256><<<rows / 16, 128, 0, stream>>>(head_struc, ws1T, bs1, ws2T, bs2, ls);
    proj_kernel<256, 256><<<rows / 16, 128, 0, stream>>>(tail_struc, ws1T, bs1, ws2T, bs2, ts);

    // one wave per (example, branch): 2*B waves / 8 waves per block
    branch_kernel<<<(2 * BATCH) / 8, 256, 0, stream>>>(lt, tt, ls, ts, perms, psum, nsum);

    finalize_kernel<<<BATCH / 256, 256, 0, stream>>>(psum, nsum, confidence, score, loss);

    logits_kernel<<<4096, 256, 0, stream>>>(tt, ts, lt, ls);
}
